// WeightedMultiplication_14508399526037
// MI455X (gfx1250) — matrix operand order repaired and checked by reading, not compiled, not run on hardware
//
#include <hip/hip_runtime.h>

// ---------------------------------------------------------------------------
// Gram(X) -> triu -> * weights -> BatchNorm, for x[16,64,256,256] f32.
// Per-slice Gram via bf16 WMMA, whole 256x256 slice resident in LDS (bf16,
// padded stride). 2x2 register blocking (32x32 super-tiles): 8 b128 LDS loads
// feed 4 v_wmma_f32_16x16x32_bf16, with double-buffered fragments so ds_loads
// for chunk k+1 overlap the wmmas of chunk k.
// ---------------------------------------------------------------------------

typedef __attribute__((ext_vector_type(16))) __bf16 v16bf;
typedef __attribute__((ext_vector_type(8)))  float  v8f;

#define H          256
#define LDS_STRIDE 264          // 256 + 8 bf16 pad -> conflict-free b128 reads
#define NPLANES    64
#define BSZ        16
#define NSLICE     (BSZ * NPLANES)   // 1024

struct Frag32B { uint4 lo, hi; };

static __device__ __forceinline__ v16bf make_frag(uint4 a, uint4 b) {
    Frag32B t{a, b};
    return __builtin_bit_cast(v16bf, t);
}

__global__ __launch_bounds__(256) void gram_bn_stats_kernel(
    const float* __restrict__ x, const float* __restrict__ w,
    float* __restrict__ out, float* __restrict__ partials)
{
    extern __shared__ unsigned short xs[];   // bf16 bits, [H][LDS_STRIDE]
    __shared__ float red[16];

    const int n   = blockIdx.x;              // slice = b*64 + c
    const int c   = n & (NPLANES - 1);       // channel
    const int tid = threadIdx.x;

    // ---- stage slice (256x256 f32 = 256KB) -> LDS bf16 (128KB) -------------
    const float4* xg = (const float4*)(x + (size_t)n * H * H);
    for (int i = tid; i < H * H / 4; i += 256) {
        float4 v = xg[i];
        int row = i >> 6;
        int col = (i & 63) << 2;
        unsigned short b0 = __builtin_bit_cast(unsigned short, (__bf16)v.x);
        unsigned short b1 = __builtin_bit_cast(unsigned short, (__bf16)v.y);
        unsigned short b2 = __builtin_bit_cast(unsigned short, (__bf16)v.z);
        unsigned short b3 = __builtin_bit_cast(unsigned short, (__bf16)v.w);
        uint2 pk;
        pk.x = ((unsigned)b1 << 16) | b0;
        pk.y = ((unsigned)b3 << 16) | b2;
        *(uint2*)&xs[row * LDS_STRIDE + col] = pk;
    }
    __syncthreads();

    const int lane = tid & 31;
    const int wave = tid >> 5;
    const int ln   = lane & 15;
    const int hi   = lane >> 4;

    float s = 0.f, s2 = 0.f;
    float*       og = out + (size_t)n * H * H;
    const float* wg = w   + (size_t)c * H * H;

    // 36 upper-tri 32x32 super-tiles over 8 waves (wave-uniform control flow)
    for (int idx = wave; idx < 36; idx += 8) {
        int si = 0, rem = idx;
        while (rem >= 8 - si) { rem -= 8 - si; ++si; }
        const int sj = si + rem;
        const int m0 = si << 5, j0 = sj << 5;

        const unsigned short* arow0 = &xs[(m0      + ln) * LDS_STRIDE];
        const unsigned short* arow1 = &xs[(m0 + 16 + ln) * LDS_STRIDE];
        const unsigned short* brow0 = &xs[(j0      + ln) * LDS_STRIDE];
        const unsigned short* brow1 = &xs[(j0 + 16 + ln) * LDS_STRIDE];

        v8f acc00 = {}, acc01 = {}, acc10 = {}, acc11 = {};

        // prologue: fragments for k0 = 0
        uint4 a00 = *(const uint4*)&arow0[hi * 8];
        uint4 a01 = *(const uint4*)&arow0[16 + hi * 8];
        uint4 a10 = *(const uint4*)&arow1[hi * 8];
        uint4 a11 = *(const uint4*)&arow1[16 + hi * 8];
        uint4 b00 = *(const uint4*)&brow0[hi * 8];
        uint4 b01 = *(const uint4*)&brow0[16 + hi * 8];
        uint4 b10 = *(const uint4*)&brow1[hi * 8];
        uint4 b11 = *(const uint4*)&brow1[16 + hi * 8];

        #pragma unroll
        for (int k0 = 0; k0 < H; k0 += 32) {
            v16bf fA0 = make_frag(a00, a01);
            v16bf fA1 = make_frag(a10, a11);
            v16bf fB0 = make_frag(b00, b01);
            v16bf fB1 = make_frag(b10, b11);

            if (k0 + 32 < H) {               // prefetch next K chunk
                const int kn = k0 + 32;
                a00 = *(const uint4*)&arow0[kn + hi * 8];
                a01 = *(const uint4*)&arow0[kn + 16 + hi * 8];
                a10 = *(const uint4*)&arow1[kn + hi * 8];
                a11 = *(const uint4*)&arow1[kn + 16 + hi * 8];
                b00 = *(const uint4*)&brow0[kn + hi * 8];
                b01 = *(const uint4*)&brow0[kn + 16 + hi * 8];
                b10 = *(const uint4*)&brow1[kn + hi * 8];
                b11 = *(const uint4*)&brow1[kn + 16 + hi * 8];
            }

            acc00 = __builtin_amdgcn_wmma_f32_16x16x32_bf16(
                false, fA0, false, fB0, (short)0, acc00, false, false);
            acc01 = __builtin_amdgcn_wmma_f32_16x16x32_bf16(
                false, fA0, false, fB1, (short)0, acc01, false, false);
            acc10 = __builtin_amdgcn_wmma_f32_16x16x32_bf16(   // discarded if si==sj
                false, fA1, false, fB0, (short)0, acc10, false, false);
            acc11 = __builtin_amdgcn_wmma_f32_16x16x32_bf16(
                false, fA1, false, fB1, (short)0, acc11, false, false);
        }

        // ---- epilogue: weight (unconditional load) + triu mask + store -----
        auto emit = [&](const v8f& acc, int r0, int c0) {
            #pragma unroll
            for (int r = 0; r < 8; ++r) {
                int row = r0 + r + (hi << 3);
                int col = c0 + ln;
                float wv = wg[row * H + col];          // always in-bounds
                float v  = (col >= row) ? acc[r] * wv : 0.f;
                og[row * H + col] = v;
                s += v; s2 += v * v;
            }
        };
        emit(acc00, m0,      j0);
        emit(acc01, m0,      j0 + 16);
        emit(acc11, m0 + 16, j0 + 16);
        if (si < sj) {
            emit(acc10, m0 + 16, j0);
            // mirrored 32x32 lower super-tile of zeros: coalesced b128 stores
            #pragma unroll
            for (int it = 0; it < 8; ++it) {
                int e   = it * 32 + lane;              // 0..255 float4 slots
                int zr  = j0 + (e >> 3);
                int zc  = m0 + ((e & 7) << 2);
                *(float4*)&og[zr * H + zc] = float4{0.f, 0.f, 0.f, 0.f};
            }
        } else {
            // diagonal super-tile: sub-tile (m0+16, j0) is strictly lower -> 0
            #pragma unroll
            for (int it = 0; it < 2; ++it) {
                int e  = it * 32 + lane;               // 0..63 float4 slots
                int zr = m0 + 16 + (e >> 2);
                int zc = j0 + ((e & 3) << 2);
                *(float4*)&og[zr * H + zc] = float4{0.f, 0.f, 0.f, 0.f};
            }
        }
    }

    // ---- block reduce (sum, sumsq) -> partials[n] (deterministic) ----------
    #pragma unroll
    for (int m = 16; m >= 1; m >>= 1) {
        s  += __shfl_xor(s,  m, 32);
        s2 += __shfl_xor(s2, m, 32);
    }
    if (lane == 0) { red[wave] = s; red[wave + 8] = s2; }
    __syncthreads();
    if (tid == 0) {
        float ts = 0.f, t2 = 0.f;
        #pragma unroll
        for (int i = 0; i < 8; ++i) { ts += red[i]; t2 += red[i + 8]; }
        partials[n * 2]     = ts;
        partials[n * 2 + 1] = t2;
    }
}

// ---------------------------------------------------------------------------
// Kernel B: BatchNorm apply, in-place on the staged G in d_out.
// ---------------------------------------------------------------------------
__global__ __launch_bounds__(256) void bn_apply_kernel(
    float* __restrict__ out, const float* __restrict__ partials,
    const float* __restrict__ gamma, const float* __restrict__ beta)
{
    const int i4    = blockIdx.x * 256 + threadIdx.x;  // float4 index
    const int slice = i4 >> 14;                        // 16384 float4 / slice
    const int c     = slice & (NPLANES - 1);

    float sp = 0.f, sq = 0.f;
    #pragma unroll
    for (int b = 0; b < BSZ; ++b) {
        sp += partials[(b * NPLANES + c) * 2];
        sq += partials[(b * NPLANES + c) * 2 + 1];
    }
    const float inv_cnt = 1.0f / ((float)BSZ * (float)(H * H));
    float mean  = sp * inv_cnt;
    float var   = sq * inv_cnt - mean * mean;
    float scale = gamma[c] * rsqrtf(var + 1e-5f);
    float bias  = beta[c] - mean * scale;

    float4 g = ((const float4*)out)[i4];
    g.x = g.x * scale + bias;
    g.y = g.y * scale + bias;
    g.z = g.z * scale + bias;
    g.w = g.w * scale + bias;
    ((float4*)out)[i4] = g;
}

// ---------------------------------------------------------------------------
extern "C" void kernel_launch(void* const* d_in, const int* in_sizes, int n_in,
                              void* d_out, int out_size, void* d_ws, size_t ws_size,
                              hipStream_t stream) {
    const float* x     = (const float*)d_in[0];   // [16,64,256,256]
    const float* w     = (const float*)d_in[1];   // [64,256,256]
    const float* gamma = (const float*)d_in[2];   // [64]
    const float* beta  = (const float*)d_in[3];   // [64]
    float* out      = (float*)d_out;              // [16,64,256,256]
    float* partials = (float*)d_ws;               // 1024 * 2 floats

    const size_t lds_bytes = (size_t)H * LDS_STRIDE * sizeof(unsigned short); // 132 KB

    gram_bn_stats_kernel<<<NSLICE, 256, lds_bytes, stream>>>(x, w, out, partials);

    const int total_f4 = BSZ * NPLANES * H * H / 4;   // 16,777,216
    bn_apply_kernel<<<total_f4 / 256, 256, 0, stream>>>(out, partials, gamma, beta);
}
